// Time_Self_Att_87488483819800
// MI455X (gfx1250) — compile-verified, matrix operand order checked by reading, not run on hardware
//
#include <hip/hip_runtime.h>

typedef __attribute__((ext_vector_type(16))) _Float16 v16h;
typedef __attribute__((ext_vector_type(8)))  _Float16 v8h;
typedef __attribute__((ext_vector_type(8)))  float    v8f;

constexpr int NB = 8;     // batch
constexpr int NC = 80;    // channels (= V width), 5 tiles of 16
constexpr int NCP = 96;   // padded V width: 5 data tiles + 1 ones tile
constexpr int NT = 2048;  // time
constexpr int ND = 256;   // K/Q feature dim, 8 chunks of 32

#define CAT16(lo, hi) __builtin_shufflevector((lo), (hi), 0,1,2,3,4,5,6,7,8,9,10,11,12,13,14,15)

// ---------------------------------------------------------------------------
// Kernel 1: K = xt @ Wk^T + bk ; Q = xt @ Wq^T + bq  (f32 math, f16 output)
// Kf/Qf layout: [B, T, 256] row-major (d contiguous).
// ---------------------------------------------------------------------------
__global__ __launch_bounds__(256) void proj_kq(
    const float* __restrict__ x, const float* __restrict__ Wk,
    const float* __restrict__ bk, const float* __restrict__ Wq,
    const float* __restrict__ bq, _Float16* __restrict__ Kf,
    _Float16* __restrict__ Qf) {
  int idx = blockIdx.x * 256 + threadIdx.x;   // t fastest -> coalesced x loads
  int t = idx & (NT - 1);
  int d = (idx >> 11) & (ND - 1);
  int b = idx >> 19;
  float aK = bk[d];
  float aQ = bq[d];
  const float* xp = x + ((size_t)b * NC) * NT + t;
  const float* wk = Wk + d * NC;
  const float* wq = Wq + d * NC;
#pragma unroll 8
  for (int c = 0; c < NC; ++c) {
    float xv = xp[(size_t)c * NT];
    aK = fmaf(xv, wk[c], aK);
    aQ = fmaf(xv, wq[c], aQ);
  }
  size_t o = ((size_t)(b * NT + t)) * ND + d;
  Kf[o] = (_Float16)aK;
  Qf[o] = (_Float16)aQ;
}

// ---------------------------------------------------------------------------
// Kernel 2: V^T in f16, padded to 96 rows: rows 0..79 = x (already [B,C,T] =
// V^T layout), rows 80..95 = 1.0 so that tile 5 of P@V yields row sums of P.
// ---------------------------------------------------------------------------
__global__ __launch_bounds__(256) void conv_v(const float* __restrict__ x,
                                              _Float16* __restrict__ Vt) {
  int i = blockIdx.x * 256 + threadIdx.x;       // over B*96*T
  int t = i & (NT - 1);
  int c = (i >> 11) % NCP;
  int b = i / (NCP * NT);
  float v = (c < NC) ? x[((size_t)(b * NC + c)) * NT + t] : 1.0f;
  Vt[i] = (_Float16)v;
}

// ---------------------------------------------------------------------------
// Kernel 3: flash attention.  One wave = one 16-row query tile.
//   S = K_tile @ Q^T / 16  -> online softmax (group max, matmul row-sums)
//   acc[0..4] += P @ V ;  acc[5] += P @ ones  (= running row sums l)
// WMMA wave32 layouts:
//   A (16x32 f16): lane = m + 16h, element e -> K = 8h + e (+8 if e>=8)
//   B (32x16 f16): lane = n + 16h, same K pattern (column n contiguous in Vt/Qf)
//   C/D (16x16 f32): reg r, lane l -> m = r + 8*(l/16), n = l%16
// ---------------------------------------------------------------------------
__global__ __launch_bounds__(128) void flash_att(
    const _Float16* __restrict__ Kf, const _Float16* __restrict__ Qf,
    const _Float16* __restrict__ Vt, float* __restrict__ out) {
  __shared__ _Float16 pshare[4 * 16 * 32];   // per-wave 16x32 P tile (f16)
  const int wave = threadIdx.x >> 5;
  const int lane = threadIdx.x & 31;
  const int n = lane & 15;    // N index (B/C/D) or A-row index
  const int h = lane >> 4;    // half-wave select
  _Float16* pb = pshare + wave * (16 * 32);

  int tile = blockIdx.x * 4 + wave;     // 0 .. B*(T/16)-1
  int b = tile >> 7;                    // 128 tiles per batch
  int i0 = (tile & 127) << 4;

  // Load the 16x256 K tile (A operands for all 8 d-chunks) once.
  const _Float16* Krow = Kf + ((size_t)(b * NT + i0 + n)) * ND + 8 * h;
  v16h Atile[8];
#pragma unroll
  for (int dc = 0; dc < 8; ++dc) {
    v8h lo = *(const v8h*)(Krow + dc * 32);
    v8h hi = *(const v8h*)(Krow + dc * 32 + 16);
    Atile[dc] = CAT16(lo, hi);
  }

  v8f acc[6] = {};                 // acc[5] = running softmax denominators l
  float m2_run = -3.0e38f;         // running (scaled, base-2) group max
  // exponent scale: (s/16) * log2(e)
  const float c1 = 0.0625f * 1.44269504088896340736f;

  for (int j0 = 0; j0 < NT; j0 += 32) {
    const _Float16* Qr0 = Qf + ((size_t)(b * NT + j0 + n)) * ND + 8 * h;
    const _Float16* Qr1 = Qr0 + (size_t)16 * ND;
    if (j0 + 32 < NT) {
      __builtin_prefetch((const void*)(Qr0 + (size_t)32 * ND), 0, 1);
      __builtin_prefetch((const void*)(Qr1 + (size_t)32 * ND), 0, 1);
    }

    // ---- S tiles: two 16x16 f32 tiles over j0..j0+31 -------------------
    v8f s0 = {}, s1 = {};
#pragma unroll
    for (int dc = 0; dc < 8; ++dc) {
      v8h lo0 = *(const v8h*)(Qr0 + dc * 32);
      v8h hi0 = *(const v8h*)(Qr0 + dc * 32 + 16);
      v16h Bq0 = CAT16(lo0, hi0);
      s0 = __builtin_amdgcn_wmma_f32_16x16x32_f16(false, Atile[dc], false, Bq0,
                                                  (short)0, s0, false, false);
      v8h lo1 = *(const v8h*)(Qr1 + dc * 32);
      v8h hi1 = *(const v8h*)(Qr1 + dc * 32 + 16);
      v16h Bq1 = CAT16(lo1, hi1);
      s1 = __builtin_amdgcn_wmma_f32_16x16x32_f16(false, Atile[dc], false, Bq1,
                                                  (short)0, s1, false, false);
    }

    // ---- online softmax, one shared max per 8-row group ----------------
    float tm = fmaxf(s0[0], s1[0]);
#pragma unroll
    for (int r = 1; r < 8; ++r) tm = fmaxf(tm, fmaxf(s0[r], s1[r]));
    tm = fmaxf(tm, __shfl_xor(tm, 1, 32));
    tm = fmaxf(tm, __shfl_xor(tm, 2, 32));
    tm = fmaxf(tm, __shfl_xor(tm, 4, 32));
    tm = fmaxf(tm, __shfl_xor(tm, 8, 32));
    float m2new = fmaxf(m2_run, tm * c1);
    float corr = __builtin_amdgcn_exp2f(m2_run - m2new);
    m2_run = m2new;

#pragma unroll
    for (int r = 0; r < 8; ++r) {
      float p0 = __builtin_amdgcn_exp2f(fmaf(s0[r], c1, -m2new));
      float p1 = __builtin_amdgcn_exp2f(fmaf(s1[r], c1, -m2new));
      int m = r + 8 * h;                      // C-layout row held by this lane
      pb[m * 32 + n]      = (_Float16)p0;     // P tile, row-major 16x32 in LDS
      pb[m * 32 + 16 + n] = (_Float16)p1;
    }
#pragma unroll
    for (int ct = 0; ct < 6; ++ct) {
#pragma unroll
      for (int r = 0; r < 8; ++r) acc[ct][r] *= corr;
    }

    // ---- reload P in A layout (wave-private LDS, DS ops in-order) ------
    const _Float16* pr = pb + n * 32 + 8 * h;  // lane's A-row = n
    v8h plo = *(const v8h*)(pr);
    v8h phi = *(const v8h*)(pr + 16);
    v16h P = CAT16(plo, phi);

    // ---- acc += P @ V : 5 data tiles + 1 ones tile (row sums) ----------
#pragma unroll
    for (int ct = 0; ct < 6; ++ct) {
      const _Float16* vp =
          Vt + ((size_t)(b * NCP + ct * 16 + n)) * NT + j0 + 8 * h;
      if (j0 + 32 < NT) __builtin_prefetch((const void*)(vp + 32), 0, 1);
      v8h vlo = *(const v8h*)(vp);
      v8h vhi = *(const v8h*)(vp + 16);
      v16h Bv = CAT16(vlo, vhi);
      acc[ct] = __builtin_amdgcn_wmma_f32_16x16x32_f16(false, P, false, Bv,
                                                       (short)0, acc[ct],
                                                       false, false);
    }
  }

  // ---- epilogue: out[b, c, i0+m] = acc / l (l = acc[5], all lanes) ------
#pragma unroll
  for (int r = 0; r < 8; ++r) {
    float invl = 1.0f / acc[5][r];
    int m = r + 8 * h;
#pragma unroll
    for (int ct = 0; ct < 5; ++ct) {
      int c = ct * 16 + n;
      out[((size_t)(b * NC + c)) * NT + i0 + m] = acc[ct][r] * invl;
    }
  }
}

// ---------------------------------------------------------------------------
extern "C" void kernel_launch(void* const* d_in, const int* in_sizes, int n_in,
                              void* d_out, int out_size, void* d_ws,
                              size_t ws_size, hipStream_t stream) {
  (void)in_sizes; (void)n_in; (void)out_size; (void)ws_size;
  const float* x  = (const float*)d_in[0];
  const float* Wk = (const float*)d_in[1];
  const float* bk = (const float*)d_in[2];
  const float* Wq = (const float*)d_in[3];
  const float* bq = (const float*)d_in[4];
  float* out = (float*)d_out;

  _Float16* Kf = (_Float16*)d_ws;                     //  8 MB
  _Float16* Qf = Kf + (size_t)NB * NT * ND;           //  8 MB
  _Float16* Vt = Qf + (size_t)NB * NT * ND;           //  3 MB (96 rows)

  proj_kq<<<(NB * NT * ND) / 256, 256, 0, stream>>>(x, Wk, bk, Wq, bq, Kf, Qf);
  conv_v<<<(NB * NCP * NT) / 256, 256, 0, stream>>>(x, Vt);
  flash_att<<<(NB * (NT / 16)) / 4, 128, 0, stream>>>(Kf, Qf, Vt, out);
}
